// Image2Seq_34935263985930
// MI455X (gfx1250) — matrix operand order repaired and checked by reading, not compiled, not run on hardware
//
#include <hip/hip_runtime.h>

// ---------------------------------------------------------------------------
// Model dims
// ---------------------------------------------------------------------------
constexpr int kB   = 32;
constexpr int kN   = 196;
constexpr int kENC = 2048;
constexpr int kV   = 32000;
constexpr int kED  = 512;
constexpr int kS   = 32;
constexpr int kH   = 1024;

typedef __attribute__((ext_vector_type(16))) __bf16 v16bf;
typedef __attribute__((ext_vector_type(2)))  __bf16 v2bf;
typedef __attribute__((ext_vector_type(8)))  float  v8f;
typedef __attribute__((ext_vector_type(4)))  unsigned int u32x4;
typedef __attribute__((ext_vector_type(8)))  unsigned int u32x8;

// packed fp32 -> bf16 pair (v_cvt_pk_bf16_f32 when available)
__device__ __forceinline__ unsigned int pack2bf(float a, float b) {
#if __has_builtin(__builtin_amdgcn_cvt_pk_bf16_f32)
    union { v2bf v; unsigned int u; } cv;
    cv.v = __builtin_amdgcn_cvt_pk_bf16_f32(a, b);
    return cv.u;
#else
    unsigned int ua = __float_as_uint(a), ub = __float_as_uint(b);
    ua = (ua + 0x7FFFu + ((ua >> 16) & 1u)) >> 16;
    ub = (ub + 0x7FFFu + ((ub >> 16) & 1u)) >> 16;
    return ua | (ub << 16);
#endif
}

union FragB { uint4 q[2]; v16bf v; };

// async global -> LDS copy of 16B, tracked by ASYNCcnt
__device__ __forceinline__ void async_copy_b128(unsigned lds_off, const void* g) {
    asm volatile("global_load_async_to_lds_b128 %0, %1, off"
                 :
                 : "v"(lds_off), "v"((unsigned long long)(uintptr_t)g)
                 : "memory");
}
__device__ __forceinline__ void async_wait0() {
    asm volatile("s_wait_asynccnt 0x0" ::: "memory");
}

// ---------------------------------------------------------------------------
// WMMA GEMM:  C[M,N] = A[M,K] (fp32) * W[N,K]^T (bf16) [+ C][+ bias1 + bias2]
// All of M%BM, N%BN, K%BK are zero at every call site, so no bounds handling
// and EXEC is all-ones (required for WMMA / async-load legality).
// TDM=true: B tile staged by the Tensor Data Mover (one descriptor per
//   k-stage, issued by wave 0; LDS row padding done by the TDM pad engine).
// TDM=false: B tile staged with per-thread async global->LDS b128 copies.
// ---------------------------------------------------------------------------
template<int BM, int BN, int BK, int WM, int WN, bool ACC, bool TDM>
__global__ __launch_bounds__((BM / WM) * (BN / WN) * 32)
void gemm_bf16(const float* __restrict__ A, int lda,
               const unsigned short* __restrict__ Bw,
               const float* __restrict__ bias1,
               const float* __restrict__ bias2,
               float* __restrict__ C, long long ldc, int K)
{
    constexpr int PAD = 8;                 // bank-conflict padding (bf16 units)
    constexpr int LDT = BK + PAD;          // 72 halves = 36 dwords per row
    __shared__ unsigned short As[BM][LDT];
    __shared__ unsigned short Bs[BN][LDT];

    constexpr int WAVES_N = BN / WN;
    constexpr int WAVES_M = BM / WM;
    constexpr int T = WAVES_M * WAVES_N * 32;
    constexpr int A_IT = (BM * (BK / 4)) / T;   // uniform, exact
    constexpr int B_IT = (BN * (BK / 8)) / T;   // uniform, exact
    constexpr int AR_STEP = T / (BK / 4);       // rows advanced per A iter
    constexpr int BR_STEP = T / (BK / 8);       // rows advanced per B iter

    const int tid   = threadIdx.x;
    const int lane  = tid & 31;
    const int wid   = tid >> 5;
    const int waveM = (wid / WAVES_N) * WM;
    const int waveN = (wid % WAVES_N) * WN;
    const int row0  = blockIdx.y * BM;
    const int col0  = blockIdx.x * BN;

    // ---- hoisted per-thread staging addresses -----------------------------
    const int ar = tid / (BK / 4), ac = (tid % (BK / 4)) * 4;
    const float* ap = A + (size_t)(row0 + ar) * lda + ac;     // advances by BK
    unsigned short* aS = &As[ar][ac];
    size_t aro[A_IT];
#pragma unroll
    for (int i = 0; i < A_IT; ++i) aro[i] = (size_t)i * AR_STEP * lda;

    const int br = tid / (BK / 8), bc = (tid % (BK / 8)) * 8;
    const unsigned short* bp = Bw + (size_t)(col0 + br) * K + bc;  // +BK/stage
    const unsigned bS = (unsigned)(uintptr_t)&Bs[br][bc];
    size_t bro[B_IT];
#pragma unroll
    for (int i = 0; i < B_IT; ++i) bro[i] = (size_t)i * BR_STEP * K;

    // ---- TDM descriptor (group1 is k-invariant) ---------------------------
    const unsigned ldsB = (unsigned)(uintptr_t)&Bs[0][0];
    const unsigned long long gB =
        (unsigned long long)(uintptr_t)(Bw + (size_t)col0 * K);
    u32x8 g1;
    {
        const unsigned kd = (unsigned)(K >> 1);        // row length in dwords
        g1[0] = (2u << 16)        // data_size = 4B units
              | (1u << 20)        // pad_enable
              | (4u << 22)        // pad_interval: 32 dwords (one tile row)
              | (3u << 25);       // pad_amount:   4 dwords (16B) -> LDT rows
        g1[1] = (kd & 0xFFFFu) << 16;                  // tensor_dim0 lo16
        g1[2] = (kd >> 16) | ((unsigned)BN << 16);     // dim0 hi16 | dim1 lo16
        g1[3] = ((unsigned)(BK / 2)) << 16;            // dim1 hi=0 | tile_dim0
        g1[4] = (unsigned)BN;                          // tile_dim1 | tile_dim2=0
        g1[5] = kd;                                    // tensor_dim0_stride lo32
        g1[6] = 0;
        g1[7] = 0;
    }

    v8f acc[WM / 16][WN / 16];
#pragma unroll
    for (int i = 0; i < WM / 16; ++i)
#pragma unroll
        for (int j = 0; j < WN / 16; ++j)
#pragma unroll
            for (int r = 0; r < 8; ++r) acc[i][j][r] = 0.0f;

    const int half0 = (lane < 16) ? 0 : 8;    // A-frag K sub-offset (ISA layout)
    const int koffB = (lane < 16) ? 0 : 8;    // B-frag K sub-offset

    for (int k0 = 0; k0 < K; k0 += BK) {
        // ---- B tile staging -----------------------------------------------
        if constexpr (TDM) {
            if (wid == 0) {
                const unsigned long long ga = gB + (unsigned long long)k0 * 2;
                u32x4 g0;
                g0[0] = 1u;                                   // count=1
                g0[1] = ldsB;                                 // lds_addr
                g0[2] = (unsigned)ga;                         // global lo32
                g0[3] = ((unsigned)(ga >> 32) & 0x01FFFFFFu)  // global hi25
                      | (2u << 30);                           // type=2 (image)
                asm volatile("tensor_load_to_lds %0, %1"
                             :
                             : "s"(g0), "s"(g1)
                             : "memory");
            }
        } else {
#pragma unroll
            for (int i = 0; i < B_IT; ++i)
                async_copy_b128(bS + (unsigned)(i * BR_STEP * LDT * 2),
                                bp + bro[i]);
            bp += BK;
        }
        // ---- A tile: fp32 global -> packed-cvt -> bf16 LDS (overlaps DMA) -
#pragma unroll
        for (int i = 0; i < A_IT; ++i) {
            const float4 f = *reinterpret_cast<const float4*>(ap + aro[i]);
            *reinterpret_cast<uint2*>(aS + i * AR_STEP * LDT) =
                make_uint2(pack2bf(f.x, f.y), pack2bf(f.z, f.w));
        }
        ap += BK;
        if (k0 + BK < K) {
#pragma unroll
            for (int i = 0; i < A_IT; ++i)
                __builtin_prefetch(ap + aro[i], 0, 1);
        }
        if constexpr (TDM) __builtin_amdgcn_s_wait_tensorcnt(0);
        else               async_wait0();
        __syncthreads();

        // ---- compute -------------------------------------------------------
#pragma unroll
        for (int kk = 0; kk < BK; kk += 32) {
            FragB fa[WM / 16], fb[WN / 16];
#pragma unroll
            for (int i = 0; i < WM / 16; ++i) {
                const int m = waveM + i * 16 + (lane & 15);
                fa[i].q[0] = *reinterpret_cast<const uint4*>(&As[m][kk + half0]);
                fa[i].q[1] = *reinterpret_cast<const uint4*>(&As[m][kk + half0 + 16]);
            }
#pragma unroll
            for (int j = 0; j < WN / 16; ++j) {
                const int n = waveN + j * 16 + (lane & 15);
                fb[j].q[0] = *reinterpret_cast<const uint4*>(&Bs[n][kk + koffB]);
                fb[j].q[1] = *reinterpret_cast<const uint4*>(&Bs[n][kk + koffB + 16]);
            }
#pragma unroll
            for (int i = 0; i < WM / 16; ++i)
#pragma unroll
                for (int j = 0; j < WN / 16; ++j)
                    acc[i][j] = __builtin_amdgcn_wmma_f32_16x16x32_bf16(
                        false, fa[i].v, false, fb[j].v, (short)0, acc[i][j],
                        false, false);
        }
        __syncthreads();
    }

    // ---- epilogue ----------------------------------------------------------
    const int rbase = (lane >> 4) << 3;   // C layout: VGPR r -> row r or r+8
#pragma unroll
    for (int i = 0; i < WM / 16; ++i) {
#pragma unroll
        for (int j = 0; j < WN / 16; ++j) {
            const int gn = col0 + waveN + j * 16 + (lane & 15);
#pragma unroll
            for (int r = 0; r < 8; ++r) {
                const int gm = row0 + waveM + i * 16 + rbase + r;
                const size_t off = (size_t)gm * (size_t)ldc + gn;
                float v = acc[i][j][r];
                if (ACC)   v += C[off];
                if (bias1) v += bias1[gn];
                if (bias2) v += bias2[gn];
                C[off] = v;
            }
        }
    }
}

// ---------------------------------------------------------------------------
// fp32 -> bf16 bulk conversion (vector-4)
// ---------------------------------------------------------------------------
__global__ void conv_bf16_4(const float* __restrict__ in,
                            unsigned short* __restrict__ out, int n4)
{
    const int i = blockIdx.x * blockDim.x + threadIdx.x;
    if (i >= n4) return;
    const float4 f = *reinterpret_cast<const float4*>(&in[(size_t)i * 4]);
    *reinterpret_cast<uint2*>(&out[(size_t)i * 4]) =
        make_uint2(pack2bf(f.x, f.y), pack2bf(f.z, f.w));
}

// ---------------------------------------------------------------------------
// mean over N + exact GELU  -> meanenc[B,ENC]
// ---------------------------------------------------------------------------
__global__ void mean_gelu(const float* __restrict__ feat, float* __restrict__ out)
{
    const int idx = blockIdx.x * blockDim.x + threadIdx.x;
    if (idx >= kB * kENC) return;
    const int b = idx >> 11, e = idx & (kENC - 1);
    float s = 0.f;
    for (int n = 0; n < kN; ++n)
        s += feat[((size_t)(b * kN + n)) * kENC + e];
    const float m = s * (1.0f / (float)kN);
    out[idx] = 0.5f * m * (1.0f + erff(m * 0.70710678118654752f));
}

// ---------------------------------------------------------------------------
// embedding gather: embeds[b,s,:] = emb[captions[b,s]]
// ---------------------------------------------------------------------------
__global__ void gather_emb(const float* __restrict__ emb, const int* __restrict__ cap,
                           float* __restrict__ out)
{
    const int idx = blockIdx.x * blockDim.x + threadIdx.x;
    if (idx >= kB * kS * kED) return;
    const int e = idx & (kED - 1);
    const int s = (idx >> 9) & (kS - 1);
    const int b = idx >> 14;
    out[idx] = emb[(size_t)cap[b * kS + s] * kED + e];
}

// ---------------------------------------------------------------------------
// attention scores + softmax; writes attn weights directly into output tensor
// one block per batch row; 8 waves, wave-parallel dot products (wave32 shfl)
// ---------------------------------------------------------------------------
__global__ void attn_softmax(const float* __restrict__ q,
                             const float* __restrict__ kbuf,
                             float* __restrict__ attnO, int t)
{
    const int b = blockIdx.x, tid = threadIdx.x;
    const int lane = tid & 31, wave = tid >> 5;
    __shared__ float ssc[224];
    __shared__ float red[256];

    const float* qp = q + (size_t)b * kH;
    for (int n = wave; n < kN; n += 8) {
        const float* kp = kbuf + ((size_t)(b * kN + n)) * kH;
        float s = 0.f;
        for (int h = lane; h < kH; h += 32) s += qp[h] * kp[h];
#pragma unroll
        for (int off = 16; off > 0; off >>= 1) s += __shfl_down(s, off, 32);
        if (lane == 0) ssc[n] = s * 0.03125f;     // 1/sqrt(1024)
    }
    __syncthreads();

    red[tid] = (tid < kN) ? ssc[tid] : -1e30f;
    __syncthreads();
    for (int st = 128; st > 0; st >>= 1) {
        if (tid < st) red[tid] = fmaxf(red[tid], red[tid + st]);
        __syncthreads();
    }
    const float mx = red[0];
    __syncthreads();

    const float e = (tid < kN) ? __expf(ssc[tid] - mx) : 0.f;
    red[tid] = e;
    __syncthreads();
    for (int st = 128; st > 0; st >>= 1) {
        if (tid < st) red[tid] += red[tid + st];
        __syncthreads();
    }
    const float inv = 1.0f / red[0];
    if (tid < kN)
        attnO[(size_t)b * kS * kN + (size_t)t * kN + tid] = e * inv;
}

// ---------------------------------------------------------------------------
// ctx_pre[b,h] = sum_n attn[b,n] * v[b,n,h]
// ---------------------------------------------------------------------------
__global__ void ctx_gather(const float* __restrict__ attnO, int t,
                           const float* __restrict__ vbuf,
                           float* __restrict__ ctxpre)
{
    const int b = blockIdx.x, tid = threadIdx.x;
    const float* ap = attnO + (size_t)b * kS * kN + (size_t)t * kN;
    float a0 = 0.f, a1 = 0.f, a2 = 0.f, a3 = 0.f;
    for (int n = 0; n < kN; ++n) {
        const float a = ap[n];
        const float4 vv = *reinterpret_cast<const float4*>(
            &vbuf[((size_t)(b * kN + n)) * kH + tid * 4]);
        a0 += a * vv.x; a1 += a * vv.y; a2 += a * vv.z; a3 += a * vv.w;
    }
    float4 o = make_float4(a0, a1, a2, a3);
    *reinterpret_cast<float4*>(&ctxpre[(size_t)b * kH + tid * 4]) = o;
}

// ---------------------------------------------------------------------------
// h = LayerNorm(h + ctx) * g + b
// ---------------------------------------------------------------------------
__global__ void add_ln(float* __restrict__ h, const float* __restrict__ ctx,
                       const float* __restrict__ g, const float* __restrict__ b2)
{
    const int b = blockIdx.x, tid = threadIdx.x;
    __shared__ float red[256];
    float x[4]; float s = 0.f;
#pragma unroll
    for (int j = 0; j < 4; ++j) {
        const int i = tid * 4 + j;
        x[j] = h[(size_t)b * kH + i] + ctx[(size_t)b * kH + i];
        s += x[j];
    }
    red[tid] = s; __syncthreads();
    for (int st = 128; st > 0; st >>= 1) { if (tid < st) red[tid] += red[tid + st]; __syncthreads(); }
    const float mu = red[0] * (1.0f / (float)kH);
    __syncthreads();
    float vs = 0.f;
#pragma unroll
    for (int j = 0; j < 4; ++j) { const float d = x[j] - mu; vs += d * d; }
    red[tid] = vs; __syncthreads();
    for (int st = 128; st > 0; st >>= 1) { if (tid < st) red[tid] += red[tid + st]; __syncthreads(); }
    const float rs = rsqrtf(red[0] * (1.0f / (float)kH) + 1e-5f);
#pragma unroll
    for (int j = 0; j < 4; ++j) {
        const int i = tid * 4 + j;
        h[(size_t)b * kH + i] = (x[j] - mu) * rs * g[i] + b2[i];
    }
}

// ---------------------------------------------------------------------------
// LSTM pointwise gate update
// ---------------------------------------------------------------------------
__global__ void lstm_step(const float* __restrict__ gates,
                          float* __restrict__ c, float* __restrict__ h)
{
    const int idx = blockIdx.x * blockDim.x + threadIdx.x;
    if (idx >= kB * kH) return;
    const int b = idx >> 10, hh = idx & (kH - 1);
    const float* gr = gates + (size_t)b * 4 * kH;
    const float ig = gr[hh], fg = gr[kH + hh], gg = gr[2 * kH + hh], og = gr[3 * kH + hh];
    const float si = 1.f / (1.f + __expf(-ig));
    const float sf = 1.f / (1.f + __expf(-fg));
    const float so = 1.f / (1.f + __expf(-og));
    const float cn = sf * c[idx] + si * tanhf(gg);
    c[idx] = cn;
    h[idx] = so * tanhf(cn);
}

// ---------------------------------------------------------------------------
// host-side orchestration
// ---------------------------------------------------------------------------
extern "C" void kernel_launch(void* const* d_in, const int* in_sizes, int n_in,
                              void* d_out, int out_size, void* d_ws, size_t ws_size,
                              hipStream_t stream)
{
    (void)in_sizes; (void)n_in; (void)out_size; (void)ws_size;

    const float* features = (const float*)d_in[0];
    const int*   captions = (const int*)  d_in[1];
    const float* emb      = (const float*)d_in[2];
    const float* q_w      = (const float*)d_in[3];
    const float* k_w      = (const float*)d_in[4];
    const float* v_w      = (const float*)d_in[5];
    const float* in_b     = (const float*)d_in[6];
    const float* out_w    = (const float*)d_in[7];
    const float* out_b    = (const float*)d_in[8];
    const float* h0_w     = (const float*)d_in[9];
    const float* h0_b     = (const float*)d_in[10];
    const float* c0_w     = (const float*)d_in[11];
    const float* c0_b     = (const float*)d_in[12];
    const float* w_ih     = (const float*)d_in[13];
    const float* b_ih     = (const float*)d_in[14];
    const float* w_hh     = (const float*)d_in[15];
    const float* b_hh     = (const float*)d_in[16];
    const float* ln_g     = (const float*)d_in[17];
    const float* ln_b2    = (const float*)d_in[18];
    const float* lin_w    = (const float*)d_in[19];
    const float* lin_b    = (const float*)d_in[20];

    float* preds = (float*)d_out;                              // [B,S,V]
    float* attnO = (float*)d_out + (size_t)kB * kS * kV;       // [B,S,N]

    // ---- workspace carve-up (256B aligned) --------------------------------
    char* ws = (char*)d_ws;
    size_t off = 0;
    auto alloc = [&](size_t bytes) -> void* {
        off = (off + 255) & ~(size_t)255;
        void* p = ws + off;
        off += bytes;
        return p;
    };
    float* kbuf    = (float*)alloc((size_t)kB * kN * kH * 4);
    float* vbuf    = (float*)alloc((size_t)kB * kN * kH * 4);
    float* meanenc = (float*)alloc((size_t)kB * kENC * 4);
    float* hbuf    = (float*)alloc((size_t)kB * kH * 4);
    float* cbuf    = (float*)alloc((size_t)kB * kH * 4);
    float* qbuf    = (float*)alloc((size_t)kB * kH * 4);
    float* ctxpre  = (float*)alloc((size_t)kB * kH * 4);
    float* ctxbuf  = (float*)alloc((size_t)kB * kH * 4);
    float* gbuf    = (float*)alloc((size_t)kB * 4 * kH * 4);
    float* embeds  = (float*)alloc((size_t)kB * kS * kED * 4);
    unsigned short* qw16   = (unsigned short*)alloc((size_t)kH * kH * 2);
    unsigned short* kw16   = (unsigned short*)alloc((size_t)kH * kENC * 2);
    unsigned short* vw16   = (unsigned short*)alloc((size_t)kH * kENC * 2);
    unsigned short* h0w16  = (unsigned short*)alloc((size_t)kH * kENC * 2);
    unsigned short* c0w16  = (unsigned short*)alloc((size_t)kH * kENC * 2);
    unsigned short* outw16 = (unsigned short*)alloc((size_t)kH * kH * 2);
    unsigned short* wih16  = (unsigned short*)alloc((size_t)4 * kH * kED * 2);
    unsigned short* whh16  = (unsigned short*)alloc((size_t)4 * kH * kH * 2);
    unsigned short* linw16 = (unsigned short*)alloc((size_t)kV * kH * 2);

    // ---- one-time: bf16 weight conversion (stays hot in 192MB L2) ---------
    auto conv = [&](const float* src, unsigned short* dst, size_t n) {
        const int n4 = (int)(n / 4);
        conv_bf16_4<<<(n4 + 255) / 256, 256, 0, stream>>>(src, dst, n4);
    };
    conv(q_w,   qw16,   (size_t)kH * kH);
    conv(k_w,   kw16,   (size_t)kH * kENC);
    conv(v_w,   vw16,   (size_t)kH * kENC);
    conv(h0_w,  h0w16,  (size_t)kH * kENC);
    conv(c0_w,  c0w16,  (size_t)kH * kENC);
    conv(out_w, outw16, (size_t)kH * kH);
    conv(w_ih,  wih16,  (size_t)4 * kH * kED);
    conv(w_hh,  whh16,  (size_t)4 * kH * kH);
    conv(lin_w, linw16, (size_t)kV * kH);

    // ---- init state --------------------------------------------------------
    mean_gelu<<<(kB * kENC) / 256, 256, 0, stream>>>(features, meanenc);
    gather_emb<<<(kB * kS * kED) / 256, 256, 0, stream>>>(emb, captions, embeds);

    // h0 / c0 : [32,2048] x [2048,1024]
    gemm_bf16<32, 128, 64, 32, 32, false, false><<<dim3(kH / 128, 1), 128, 0, stream>>>(
        meanenc, kENC, h0w16, h0_b, nullptr, hbuf, kH, kENC);
    gemm_bf16<32, 128, 64, 32, 32, false, false><<<dim3(kH / 128, 1), 128, 0, stream>>>(
        meanenc, kENC, c0w16, c0_b, nullptr, cbuf, kH, kENC);

    // K / V projections : [6272,2048] x [2048,1024]  (TDM-staged B tiles)
    gemm_bf16<64, 64, 64, 32, 32, false, true><<<dim3(kH / 64, (kB * kN) / 64), 128, 0, stream>>>(
        features, kENC, kw16, in_b + kH, nullptr, kbuf, kH, kENC);
    gemm_bf16<64, 64, 64, 32, 32, false, true><<<dim3(kH / 64, (kB * kN) / 64), 128, 0, stream>>>(
        features, kENC, vw16, in_b + 2 * kH, nullptr, vbuf, kH, kENC);

    // ---- sequential decode -------------------------------------------------
    for (int t = 0; t < kS; ++t) {
        // q = h @ q_w.T + in_b[:H]
        gemm_bf16<32, 128, 64, 32, 32, false, false><<<dim3(kH / 128, 1), 128, 0, stream>>>(
            hbuf, kH, qw16, in_b, nullptr, qbuf, kH, kH);
        // attention weights (written straight to output tensor)
        attn_softmax<<<kB, 256, 0, stream>>>(qbuf, kbuf, attnO, t);
        // context = attn @ v
        ctx_gather<<<kB, 256, 0, stream>>>(attnO, t, vbuf, ctxpre);
        // ctx = ctx_pre @ out_w.T + out_b
        gemm_bf16<32, 128, 64, 32, 32, false, false><<<dim3(kH / 128, 1), 128, 0, stream>>>(
            ctxpre, kH, outw16, out_b, nullptr, ctxbuf, kH, kH);
        // h = LN(h + ctx)
        add_ln<<<kB, 256, 0, stream>>>(hbuf, ctxbuf, ln_g, ln_b2);
        // gates = x_emb @ w_ih.T + (b_ih + b_hh)   then  += h @ w_hh.T
        gemm_bf16<32, 128, 64, 32, 32, false, false><<<dim3(4 * kH / 128, 1), 128, 0, stream>>>(
            embeds + (size_t)t * kED, kS * kED, wih16, b_ih, b_hh, gbuf, 4 * kH, kED);
        gemm_bf16<32, 128, 64, 32, 32, true, false><<<dim3(4 * kH / 128, 1), 128, 0, stream>>>(
            hbuf, kH, whh16, nullptr, nullptr, gbuf, 4 * kH, kH);
        // LSTM pointwise
        lstm_step<<<(kB * kH) / 256, 256, 0, stream>>>(gbuf, cbuf, hbuf);
        // logits: h @ lin_w.T + lin_b  -> preds[:, t, :]  (ldc = S*V)
        gemm_bf16<32, 128, 64, 32, 32, false, false><<<dim3(kV / 128, 1), 128, 0, stream>>>(
            hbuf, kH, linw16, lin_b, nullptr,
            preds + (size_t)t * kV, (long long)kS * kV, kH);
    }
}
